// DecoderBlockV2_88158498718103
// MI455X (gfx1250) — compile-verified, matrix operand order checked by reading, not run on hardware
//
#include <hip/hip_runtime.h>
#include <hip/hip_bf16.h>
#include <math.h>

// ---------------- problem constants ----------------
#define DM      512          // D_MODEL
#define DI      1024         // D_INNER
#define DS      16           // D_STATE
#define DTR     32           // DT_RANK
#define NB      64           // batch
#define TLOW    64
#define TSKIP   128
#define NT      (NB*TSKIP)   // 8192 tokens
#define NTL     (NB*TLOW)    // 4096 tokens

typedef __attribute__((ext_vector_type(16))) __bf16 v16bf;
typedef __attribute__((ext_vector_type(8)))  float  v8f;

union FragAB { v16bf bf; int4 q[2]; unsigned short us[16]; };

// packed fp32x2 -> bf16x2 via the gfx1250 packed convert (RNE), no repack fixups
__device__ inline unsigned int f2bf_pk(float a, float b) {
  unsigned int r;
  asm("v_cvt_pk_bf16_f32 %0, %1, %2" : "=v"(r) : "v"(a), "v"(b));
  return r;
}
__device__ inline uint2 f2bf_x4(float4 f) {
  uint2 r; r.x = f2bf_pk(f.x, f.y); r.y = f2bf_pk(f.z, f.w); return r;
}
__device__ inline float gelu_f(float x) { return 0.5f * x * (1.0f + erff(x * 0.70710678118f)); }
__device__ inline float silu_f(float x) { return x / (1.0f + __expf(-x)); }
__device__ inline float softplus_f(float x) { return (x > 20.f) ? x : log1pf(__expf(x)); }

#define BN_SCALE 0.9999950000374997f   /* 1/sqrt(1+1e-5), BatchNorm eval with rm=0, rv=1 */

// =====================================================================
// WMMA bf16 GEMM:  C[M,N] = epi( A[M,K] * W^T + bias )  (+ optional Cin)
//   A: fp32 row-major, lda.  W element (n,k) at W[n*wns + k*wks] (fp32).
//   fp32 -> bf16 (v_cvt_pk_bf16_f32) while staging into double-buffered LDS.
//   REMAP=1: per-batch shifted A rows (a_off) + strided output scatter
//            (out_row = b*tout + 2*(m%tin) + rpar)  -> ConvTranspose1d.
// Block: 256 threads = 8 waves (4x2).  Wave tile: 32 x (BNt/2).
// Pipeline: global loads of tile k+1 overlap WMMAs on tile k; one
// workgroup barrier per K-step.
// =====================================================================
template<int BM, int BNt, int EPI, int REMAP>
__global__ __launch_bounds__(256)
void gemm_wmma(const float* __restrict__ A, int lda,
               const float* __restrict__ W, int wns, int wks,
               const float* __restrict__ bias,
               const float* __restrict__ Cin,
               float* __restrict__ Cout, int ldc,
               int M, int N, int K,
               int tin, int tout, int rpar, int a_off)
{
  constexpr int KT  = 32;
  constexpr int LDW = 40;             // ushort row stride: 80B -> conflict-free b128 frag loads
  constexpr int WM  = BM / 4;         // 32
  constexpr int WN  = BNt / 2;        // 64 or 32
  constexpr int FM  = WM / 16;        // 2
  constexpr int FN  = WN / 16;        // 4 or 2
  constexpr int A4  = (BM  * KT) / (4 * 256);   // float4 loads/thread for A tile
  constexpr int B4  = (BNt * KT) / (4 * 256);   // float4 loads/thread for W tile

  __shared__ unsigned short As[2][BM  * LDW];
  __shared__ unsigned short Bs[2][BNt * LDW];

  const int tid  = threadIdx.x;
  const int lane = tid & 31;
  const int wid  = tid >> 5;
  const int wm0  = (wid >> 1) * WM;
  const int wn0  = (wid & 1)  * WN;
  const int bm0  = blockIdx.y * BM;
  const int bn0  = blockIdx.x * BNt;

  auto out_row = [&](int m) -> int {
    if (!REMAP) return m;
    int bb = m / tin; int j = m - bb * tin;
    return bb * tout + 2 * j + rpar;
  };

  float4 fa[A4], fb[B4];

  auto stage_load = [&](int k0) {
    #pragma unroll
    for (int i = 0; i < A4; i++) {
      int e = tid + i * 256;
      int r = e >> 3;               // 8 float4 per 32-elem row
      int c = (e & 7) << 2;
      int m = bm0 + r;
      if (REMAP) {
        int bb = m / tin; int j = m - bb * tin + a_off;
        if (j >= 0 && j < tin)
          fa[i] = *(const float4*)(A + (size_t)(bb * tin + j) * lda + k0 + c);
        else
          fa[i] = make_float4(0.f, 0.f, 0.f, 0.f);
      } else {
        fa[i] = *(const float4*)(A + (size_t)m * lda + k0 + c);
      }
    }
    if (wks == 1) {                 // contiguous K: vector loads
      #pragma unroll
      for (int i = 0; i < B4; i++) {
        int e = tid + i * 256;
        int n = e >> 3, c = (e & 7) << 2;
        fb[i] = *(const float4*)(W + (size_t)(bn0 + n) * wns + k0 + c);
      }
    } else {                        // strided weights (ConvTranspose)
      #pragma unroll
      for (int i = 0; i < B4; i++) {
        int e = tid + i * 256;
        int n = e >> 3, c = (e & 7) << 2;
        const float* p = W + (size_t)(bn0 + n) * wns + (size_t)(k0 + c) * wks;
        fb[i].x = p[0]; fb[i].y = p[wks]; fb[i].z = p[2 * wks]; fb[i].w = p[3 * wks];
      }
    }
  };

  auto stage_store = [&](int buf) {
    #pragma unroll
    for (int i = 0; i < A4; i++) {
      int e = tid + i * 256;
      int r = e >> 3, c = (e & 7) << 2;
      *(uint2*)&As[buf][r * LDW + c] = f2bf_x4(fa[i]);   // 2x v_cvt_pk_bf16_f32 + ds_store_b64
    }
    #pragma unroll
    for (int i = 0; i < B4; i++) {
      int e = tid + i * 256;
      int n = e >> 3, c = (e & 7) << 2;
      *(uint2*)&Bs[buf][n * LDW + c] = f2bf_x4(fb[i]);
    }
  };

  v8f acc[FM][FN];
  #pragma unroll
  for (int i = 0; i < FM; i++)
    #pragma unroll
    for (int j = 0; j < FN; j++)
      #pragma unroll
      for (int e = 0; e < 8; e++) acc[i][j][e] = 0.f;

  if (Cin) {   // accumulate into existing output (split-K halves / conv-transpose taps)
    #pragma unroll
    for (int fm = 0; fm < FM; fm++)
      #pragma unroll
      for (int fn = 0; fn < FN; fn++) {
        int n  = bn0 + wn0 + fn * 16 + (lane & 15);
        int mb = bm0 + wm0 + fm * 16 + ((lane >> 4) << 3);
        #pragma unroll
        for (int e = 0; e < 8; e++)
          acc[fm][fn][e] = Cin[(size_t)out_row(mb + e) * ldc + n];
      }
  }

  // ---- pipelined main loop ----
  stage_load(0);
  stage_store(0);
  __syncthreads();
  int cur = 0;
  for (int k0 = 0; k0 < K; k0 += KT) {
    const bool more = (k0 + KT) < K;
    if (more) stage_load(k0 + KT);          // global loads overlap WMMAs below

    const int idx = lane & 15;
    const int kh  = (lane >> 4) << 3;       // lanes 0-15: K 0-7,16-23 ; 16-31: K 8-15,24-31
    FragAB aF[FM], bF[FN];
    #pragma unroll
    for (int fm = 0; fm < FM; fm++) {
      const unsigned short* p = &As[cur][(wm0 + fm * 16 + idx) * LDW + kh];
      aF[fm].q[0] = *(const int4*)p;
      aF[fm].q[1] = *(const int4*)(p + 16);
    }
    #pragma unroll
    for (int fn = 0; fn < FN; fn++) {
      const unsigned short* p = &Bs[cur][(wn0 + fn * 16 + idx) * LDW + kh];
      bF[fn].q[0] = *(const int4*)p;
      bF[fn].q[1] = *(const int4*)(p + 16);
    }
    #pragma unroll
    for (int fm = 0; fm < FM; fm++)
      #pragma unroll
      for (int fn = 0; fn < FN; fn++)
        acc[fm][fn] = __builtin_amdgcn_wmma_f32_16x16x32_bf16(
            false, aF[fm].bf, false, bF[fn].bf, (short)0, acc[fm][fn], false, false);

    if (more) stage_store(cur ^ 1);
    __syncthreads();
    cur ^= 1;
  }

  // ---- epilogue: bias + activation + (remapped) store ----
  #pragma unroll
  for (int fm = 0; fm < FM; fm++)
    #pragma unroll
    for (int fn = 0; fn < FN; fn++) {
      int n  = bn0 + wn0 + fn * 16 + (lane & 15);
      float bv = bias ? bias[n] : 0.f;
      int mb = bm0 + wm0 + fm * 16 + ((lane >> 4) << 3);
      #pragma unroll
      for (int e = 0; e < 8; e++) {
        float v = acc[fm][fn][e] + bv;
        if (EPI == 1) v = gelu_f(v);
        else if (EPI == 2) v = softplus_f(v);
        else if (EPI == 3) v = silu_f(v);
        Cout[(size_t)out_row(mb + e) * ldc + n] = v;
      }
    }
}

// =====================================================================
// LayerNorm (wave per row):  out = act(LN(x1*stok + x2)) + addo
// =====================================================================
template<int ACT, int D>
__global__ __launch_bounds__(256)
void ln_k(const float* __restrict__ x1, const float* __restrict__ x2,
          const float* __restrict__ stok,
          const float* __restrict__ g, const float* __restrict__ b,
          const float* __restrict__ addo, float* __restrict__ out, int nrows)
{
  constexpr int PER = D / 32;
  int lane = threadIdx.x & 31, wid = threadIdx.x >> 5;
  int row = blockIdx.x * 8 + wid;
  if (row >= nrows) return;
  size_t base = (size_t)row * D;
  float sc = stok ? stok[row] : 1.f;
  float v[PER]; float sum = 0.f, sq = 0.f;
  #pragma unroll
  for (int j = 0; j < PER; j++) {
    int e = lane + (j << 5);
    float t = x1[base + e] * sc;
    if (x2) t += x2[base + e];
    v[j] = t; sum += t; sq += t * t;
  }
  for (int m = 16; m; m >>= 1) { sum += __shfl_xor(sum, m, 32); sq += __shfl_xor(sq, m, 32); }
  float mean = sum * (1.f / D);
  float var  = sq  * (1.f / D) - mean * mean;
  float rstd = rsqrtf(var + 1e-5f);
  #pragma unroll
  for (int j = 0; j < PER; j++) {
    int e = lane + (j << 5);
    float t = (v[j] - mean) * rstd * g[e] + b[e];
    if (ACT == 1) t = gelu_f(t);
    if (addo) t += addo[base + e];
    out[base + e] = t;
  }
}

// attention-gate psi: att = sigmoid( relu(t1_row) . psi_w + psi_b ), wave/row
__global__ __launch_bounds__(256)
void psi_k(const float* __restrict__ t1, const float* __restrict__ pw,
           const float* __restrict__ pb, float* __restrict__ att)
{
  int lane = threadIdx.x & 31, wid = threadIdx.x >> 5;
  int row = blockIdx.x * 8 + wid;
  if (row >= NT) return;
  float s = 0.f;
  #pragma unroll
  for (int j = 0; j < DM / 32; j++) {
    int e = lane + (j << 5);
    float v = t1[(size_t)row * DM + e];
    v = v > 0.f ? v : 0.f;
    s += v * pw[e];
  }
  for (int m = 16; m; m >>= 1) s += __shfl_xor(s, m, 32);
  if (!lane) att[row] = 1.f / (1.f + __expf(-(s + pb[0])));
}

// ConvTranspose post: xu = gelu( (xct + ct_b) * bn_g/sqrt(1+eps) + bn_b )
__global__ __launch_bounds__(256)
void upsample_post_k(const float* __restrict__ xct, const float* __restrict__ cb,
                     const float* __restrict__ bg, const float* __restrict__ bb,
                     float* __restrict__ xu)
{
  int i = blockIdx.x * 256 + threadIdx.x;
  if (i >= NT * DM) return;
  int d = i & (DM - 1);
  float v = (xct[i] + cb[d]) * (bg[d] * BN_SCALE) + bb[d];
  xu[i] = gelu_f(v);
}

// Mamba depthwise causal conv (k=4) + SiLU.  dir=1 -> time-reversed (backward branch).
__global__ __launch_bounds__(256)
void mamba_conv_k(const float* __restrict__ xz, const float* __restrict__ cw,
                  const float* __restrict__ cb, float* __restrict__ xs, int dir)
{
  int i = blockIdx.x * 256 + threadIdx.x;
  if (i >= NT * DI) return;
  int c = i & (DI - 1); int tok = i >> 10;
  int b = tok >> 7, t = tok & 127;
  float acc = cb[c];
  #pragma unroll
  for (int k = 0; k < 4; k++) {
    int tt = dir ? (t + 3 - k) : (t - 3 + k);
    if (tt >= 0 && tt < TSKIP)
      acc += cw[c * 4 + k] * xz[(size_t)(b * TSKIP + tt) * (2 * DI) + c];
  }
  xs[i] = silu_f(acc);
}

// Selective scan: thread = (batch, channel); 16 states in registers, 128 serial steps.
__global__ __launch_bounds__(256)
void scan_k(const float* __restrict__ delta, const float* __restrict__ xs,
            const float* __restrict__ dbl, const float* __restrict__ A_log,
            float* __restrict__ Y, int dir)
{
  int idx = blockIdx.x * 256 + threadIdx.x;      // 0 .. NB*DI-1
  if (idx >= NB * DI) return;
  int c = idx & (DI - 1); int b = idx >> 10;
  float Ar[DS];
  #pragma unroll
  for (int s = 0; s < DS; s++) Ar[s] = -__expf(A_log[c * DS + s]);
  float h[DS];
  #pragma unroll
  for (int s = 0; s < DS; s++) h[s] = 0.f;
  for (int i = 0; i < TSKIP; i++) {
    int t = dir ? (TSKIP - 1 - i) : i;
    int tok = b * TSKIP + t;
    float d  = delta[(size_t)tok * DI + c];
    float xv = xs[(size_t)tok * DI + c];
    const float* bc = dbl + (size_t)tok * (DTR + 2 * DS);
    float dx = d * xv, y = 0.f;
    #pragma unroll
    for (int s = 0; s < DS; s++) {
      float hb = h[s] * __expf(d * Ar[s]) + dx * bc[DTR + s];
      h[s] = hb;
      y += hb * bc[DTR + DS + s];
    }
    Y[(size_t)tok * DI + c] = y;
  }
}

// y = (ys + xs*D) * silu(z)
__global__ __launch_bounds__(256)
void yfuse_k(float* __restrict__ y, const float* __restrict__ xs,
             const float* __restrict__ Dp, const float* __restrict__ xz)
{
  int i = blockIdx.x * 256 + threadIdx.x;
  if (i >= NT * DI) return;
  int c = i & (DI - 1); int tok = i >> 10;
  float z = xz[(size_t)tok * (2 * DI) + DI + c];
  y[i] = (y[i] + xs[i] * Dp[c]) * silu_f(z);
}

// FFN depthwise conv (k=3, same pad) + BN(eval) + GELU
__global__ __launch_bounds__(256)
void ffn_dw_k(const float* __restrict__ in, const float* __restrict__ w,
              const float* __restrict__ wb, const float* __restrict__ bg,
              const float* __restrict__ bb, float* __restrict__ out)
{
  int i = blockIdx.x * 256 + threadIdx.x;
  if (i >= NT * DM) return;
  int d = i & (DM - 1); int tok = i >> 9;
  int b = tok >> 7, t = tok & 127;
  float acc = wb[d];
  #pragma unroll
  for (int k = 0; k < 3; k++) {
    int tt = t - 1 + k;
    if (tt >= 0 && tt < TSKIP)
      acc += w[d * 3 + k] * in[(size_t)(b * TSKIP + tt) * DM + d];
  }
  out[i] = gelu_f(acc * (bg[d] * BN_SCALE) + bb[d]);
}

// =====================================================================
extern "C" void kernel_launch(void* const* d_in, const int* in_sizes, int n_in,
                              void* d_out, int out_size, void* d_ws, size_t ws_size,
                              hipStream_t stream)
{
  (void)in_sizes; (void)n_in; (void)out_size; (void)ws_size;
  const float* x      = (const float*)d_in[0];
  const float* skip   = (const float*)d_in[1];
  const float* ct_w   = (const float*)d_in[2];
  const float* ct_b   = (const float*)d_in[3];
  const float* ct_bng = (const float*)d_in[4];
  const float* ct_bnb = (const float*)d_in[5];
  const float* ag_wg  = (const float*)d_in[6];
  const float* ag_wx  = (const float*)d_in[7];
  const float* ag_pw  = (const float*)d_in[8];
  const float* ag_pb  = (const float*)d_in[9];
  const float* ag_lg  = (const float*)d_in[10];
  const float* ag_lb  = (const float*)d_in[11];
  const float* fu1_w  = (const float*)d_in[12];
  const float* fu1_b  = (const float*)d_in[13];
  const float* fl1_g  = (const float*)d_in[14];
  const float* fl1_b  = (const float*)d_in[15];
  const float* fu2_w  = (const float*)d_in[16];
  const float* fu2_b  = (const float*)d_in[17];
  const float* fl2_g  = (const float*)d_in[18];
  const float* fl2_b  = (const float*)d_in[19];
  const float* n1_g   = (const float*)d_in[20];
  const float* n1_b   = (const float*)d_in[21];
  const float* n2_g   = (const float*)d_in[22];
  const float* n2_b   = (const float*)d_in[23];
  const float* mp[2][9];
  for (int i = 0; i < 9; i++) { mp[0][i] = (const float*)d_in[24 + i]; mp[1][i] = (const float*)d_in[33 + i]; }
  const float* bim_w  = (const float*)d_in[42];
  const float* bim_b  = (const float*)d_in[43];
  const float* bim_lg = (const float*)d_in[44];
  const float* bim_lb = (const float*)d_in[45];
  const float* dw_w   = (const float*)d_in[46];
  const float* dw_b   = (const float*)d_in[47];
  const float* fbn_g  = (const float*)d_in[48];
  const float* fbn_b  = (const float*)d_in[49];
  const float* m1_w   = (const float*)d_in[50];
  const float* m1_b   = (const float*)d_in[51];
  const float* m2_w   = (const float*)d_in[52];
  const float* m2_b   = (const float*)d_in[53];
  const float* fln_g  = (const float*)d_in[54];
  const float* fln_b  = (const float*)d_in[55];
  float* out = (float*)d_out;

  // workspace arenas (floats), reused by lifetime; peak ~260 MB
  float* WS    = (float*)d_ws;
  float* XCT   = WS + 0;          // 4M  : upsample pre-act    -> T1 -> BIMH
  float* XU    = WS + 4194304;    // 4M  : upsampled tokens    -> OUT0 -> M2O
  float* GATED = WS + 8388608;    // 4M  : gated skip          -> H2 -> RES2
  float* H1    = WS + 12582912;   // 8M  : fu1 pre-LN          -> DELTA
  float* FUSED = WS + 20971520;   // 8M  : fused tokens        -> Y (scan out)
  float* RES1  = WS + 29360128;   // 4M  : first residual
  float* XZ    = WS + 33554432;   // 16M : mamba in_proj out   -> HID
  float* XS    = WS + 50331648;   // 8M  : conv+silu out
  float* DBL   = WS + 58720256;   // 512K: x_proj out (dt|B|C)
  float* MF    = WS + 59244544;   // 4M  : fwd mamba out       -> OUT2
  float* MB    = WS + 63438848;   // 4M  : bwd mamba out       -> CBUF
  float* ATT   = WS + 67633152;   // 8K  : per-token gate
  float* T1 = XCT, *BIMH = XCT, *OUT0 = XU, *M2O = XU;
  float* H2 = GATED, *RES2 = GATED, *DELTA = H1, *Y = FUSED;
  float* HID = XZ, *OUT2 = MF, *CBUF = MB;

  const dim3 blk(256);
  const dim3 gUP(DM / 128, NTL / 128);       // (4, 32)
  const dim3 g512(DM / 128, NT / 128);       // (4, 64)
  const dim3 g1024(2 * DM / 128, NT / 128);  // (8, 64)
  const dim3 g2048(4 * DM / 128, NT / 128);  // (16, 64)
  const int  EW512  = (NT * DM) / 256;       // 16384
  const int  EW1024 = (NT * DI) / 256;       // 32768
  const int  ROWB   = NT / 8;                // 1024

  // ---- upsample: ConvTranspose1d(k=4,s=2,p=1) as 4 shifted GEMM taps ----
  // even t=2j: in[j]@Wk1 + in[j-1]@Wk3 ; odd t=2j+1: in[j+1]@Wk0 + in[j]@Wk2
  gemm_wmma<128,128,0,1><<<gUP, blk, 0, stream>>>(x, DM, ct_w + 1, 4, 2048, nullptr, nullptr, XCT, DM, NTL, DM, DM, TLOW, TSKIP, 0,  0);
  gemm_wmma<128,128,0,1><<<gUP, blk, 0, stream>>>(x, DM, ct_w + 3, 4, 2048, nullptr, XCT,     XCT, DM, NTL, DM, DM, TLOW, TSKIP, 0, -1);
  gemm_wmma<128,128,0,1><<<gUP, blk, 0, stream>>>(x, DM, ct_w + 0, 4, 2048, nullptr, nullptr, XCT, DM, NTL, DM, DM, TLOW, TSKIP, 1, +1);
  gemm_wmma<128,128,0,1><<<gUP, blk, 0, stream>>>(x, DM, ct_w + 2, 4, 2048, nullptr, XCT,     XCT, DM, NTL, DM, DM, TLOW, TSKIP, 1,  0);
  upsample_post_k<<<EW512, blk, 0, stream>>>(XCT, ct_b, ct_bng, ct_bnb, XU);

  // ---- attention gate ----
  gemm_wmma<128,128,0,0><<<g512, blk, 0, stream>>>(XU,   DM, ag_wg, DM, 1, nullptr, nullptr, T1, DM, NT, DM, DM, 0,0,0,0);
  gemm_wmma<128,128,0,0><<<g512, blk, 0, stream>>>(skip, DM, ag_wx, DM, 1, nullptr, T1,      T1, DM, NT, DM, DM, 0,0,0,0);
  psi_k<<<ROWB, blk, 0, stream>>>(T1, ag_pw, ag_pb, ATT);
  ln_k<0, DM><<<ROWB, blk, 0, stream>>>(skip, nullptr, ATT, ag_lg, ag_lb, nullptr, GATED, NT);

  // ---- fusion MLP (concat handled as two K-half GEMMs) ----
  gemm_wmma<128,128,0,0><<<g1024, blk, 0, stream>>>(XU,    DM, fu1_w,      2*DM, 1, nullptr, nullptr, H1, 2*DM, NT, 2*DM, DM, 0,0,0,0);
  gemm_wmma<128,128,0,0><<<g1024, blk, 0, stream>>>(GATED, DM, fu1_w + DM, 2*DM, 1, fu1_b,   H1,      H1, 2*DM, NT, 2*DM, DM, 0,0,0,0);
  ln_k<1, 2*DM><<<ROWB, blk, 0, stream>>>(H1, nullptr, nullptr, fl1_g, fl1_b, nullptr, FUSED, NT);
  gemm_wmma<128,128,0,0><<<g512, blk, 0, stream>>>(FUSED, 2*DM, fu2_w, 2*DM, 1, fu2_b, nullptr, H2, DM, NT, DM, 2*DM, 0,0,0,0);
  ln_k<0, DM><<<ROWB, blk, 0, stream>>>(H2,   nullptr, nullptr, fl2_g, fl2_b, nullptr, RES1, NT);
  ln_k<0, DM><<<ROWB, blk, 0, stream>>>(RES1, nullptr, nullptr, n1_g,  n1_b,  nullptr, OUT0, NT);

  // ---- bidirectional Mamba (dir=1 processes reversed time in-place) ----
  for (int dir = 0; dir < 2; dir++) {
    const float* const* p = mp[dir];   // in_proj, conv_w, conv_b, x_proj, dt_w, dt_b, A_log, D, out_w
    gemm_wmma<128,128,0,0><<<g2048, blk, 0, stream>>>(OUT0, DM, p[0], DM, 1, nullptr, nullptr, XZ, 2*DI, NT, 2*DI, DM, 0,0,0,0);
    mamba_conv_k<<<EW1024, blk, 0, stream>>>(XZ, p[1], p[2], XS, dir);
    gemm_wmma<128, 64,0,0><<<dim3(1, NT/128), blk, 0, stream>>>(XS, DI, p[3], DI, 1, nullptr, nullptr, DBL, DTR + 2*DS, NT, 64, DI, 0,0,0,0);
    gemm_wmma<128,128,2,0><<<dim3(DI/128, NT/128), blk, 0, stream>>>(DBL, DTR + 2*DS, p[4], DTR, 1, p[5], nullptr, DELTA, DI, NT, DI, DTR, 0,0,0,0);
    scan_k<<<(NB * DI) / 256, blk, 0, stream>>>(DELTA, XS, DBL, p[6], Y, dir);
    yfuse_k<<<EW1024, blk, 0, stream>>>(Y, XS, p[7], XZ);
    gemm_wmma<128,128,0,0><<<g512, blk, 0, stream>>>(Y, DI, p[8], DI, 1, nullptr, nullptr, dir ? MB : MF, DM, NT, DM, DI, 0,0,0,0);
  }

  // ---- bidirectional combine + residual ----
  gemm_wmma<128,128,0,0><<<g512, blk, 0, stream>>>(MF, DM, bim_w,      2*DM, 1, nullptr, nullptr, BIMH, DM, NT, DM, DM, 0,0,0,0);
  gemm_wmma<128,128,0,0><<<g512, blk, 0, stream>>>(MB, DM, bim_w + DM, 2*DM, 1, bim_b,   BIMH,    BIMH, DM, NT, DM, DM, 0,0,0,0);
  ln_k<1, DM><<<ROWB, blk, 0, stream>>>(BIMH, nullptr, nullptr, bim_lg, bim_lb, RES1, RES2, NT);
  ln_k<0, DM><<<ROWB, blk, 0, stream>>>(RES2, nullptr, nullptr, n2_g,   n2_b,   nullptr, OUT2, NT);

  // ---- ConvFFN: depthwise branch + MLP branch, LN, final residual ----
  ffn_dw_k<<<EW512, blk, 0, stream>>>(OUT2, dw_w, dw_b, fbn_g, fbn_b, CBUF);
  gemm_wmma<128,128,1,0><<<g2048, blk, 0, stream>>>(OUT2, DM,  m1_w, DM,  1, m1_b, nullptr, HID, 4*DM, NT, 4*DM, DM, 0,0,0,0);
  gemm_wmma<128,128,0,0><<<g512,  blk, 0, stream>>>(HID, 4*DM, m2_w, 4*DM,1, m2_b, nullptr, M2O, DM,   NT, DM, 4*DM, 0,0,0,0);
  ln_k<0, DM><<<ROWB, blk, 0, stream>>>(CBUF, M2O, nullptr, fln_g, fln_b, RES2, out, NT);
}
